// Transformer_51135880626893
// MI455X (gfx1250) — compile-verified, matrix operand order checked by reading, not run on hardware
//
#include <hip/hip_runtime.h>

#define D_MODEL 256
#define NHEAD 8
#define HDIM 32
#define BATCH 8
#define L_ENC 1024
#define N_Q 80
#define FFDIM 2048
#define LN_EPSF 1e-5f

typedef __attribute__((ext_vector_type(16))) __bf16    v16bf;
typedef __attribute__((ext_vector_type(8)))  __bf16    v8bf;
typedef __attribute__((ext_vector_type(8)))  float     v8f;
typedef __attribute__((ext_vector_type(4)))  float     v4f;
typedef unsigned short u16;
typedef __attribute__((ext_vector_type(8))) u16        v8u16;
typedef __attribute__((ext_vector_type(8))) unsigned   v8u32;
typedef __attribute__((ext_vector_type(4))) unsigned   v4u32;

// ---------- bf16 helpers (round-to-nearest-even, pure integer ops) ----------
__device__ __forceinline__ u16 f2bf(float f) {
  unsigned u = __builtin_bit_cast(unsigned, f);
  u += 0x7fffu + ((u >> 16) & 1u);
  return (u16)(u >> 16);
}
__device__ __forceinline__ unsigned pack2(float a, float b) {
  return (unsigned)f2bf(a) | ((unsigned)f2bf(b) << 16);
}
__device__ __forceinline__ __bf16 bfbits(u16 u) {
  return __builtin_bit_cast(__bf16, u);
}

// Fragment gather for V_WMMA_F32_16X16X32_BF16 A/B operands.
// Per CDNA5 ISA 7.12.2 (16-bit A 16x32): lane&15 = row, half-wave selects
// K base hb (0/8); elements 0..7 -> K=hb+0..7, elements 8..15 -> K=hb+16..23.
__device__ __forceinline__ v16bf frag_from_lds_row(const u16* row, int hb) {
  v8bf lo = __builtin_bit_cast(v8bf, *(const v8u16*)(row + hb));
  v8bf hi = __builtin_bit_cast(v8bf, *(const v8u16*)(row + hb + 16));
  return __builtin_shufflevector(lo, hi, 0, 1, 2, 3, 4, 5, 6, 7,
                                 8, 9, 10, 11, 12, 13, 14, 15);
}
__device__ __forceinline__ v16bf frag_from_global_row(const float* p, int hb) {
  v4f x0 = *(const v4f*)(p + hb);
  v4f x1 = *(const v4f*)(p + hb + 4);
  v4f x2 = *(const v4f*)(p + hb + 16);
  v4f x3 = *(const v4f*)(p + hb + 20);
  v8u32 t;
  t[0] = pack2(x0[0], x0[1]); t[1] = pack2(x0[2], x0[3]);
  t[2] = pack2(x1[0], x1[1]); t[3] = pack2(x1[2], x1[3]);
  t[4] = pack2(x2[0], x2[1]); t[5] = pack2(x2[2], x2[3]);
  t[6] = pack2(x3[0], x3[1]); t[7] = pack2(x3[2], x3[3]);
  return __builtin_bit_cast(v16bf, t);
}

// ---------------- GEMM: Y[rows,cols] = X[rows,K] @ W[cols,K]^T + bias -------
// 128x128 block tile, 8 waves (4 row x 2 col), each wave 32x64 = 8 WMMAs/step.
__launch_bounds__(256)
__global__ void gemm_bias_kernel(const float* __restrict__ X, int ldx,
                                 const float* __restrict__ W, int ldw,
                                 const float* __restrict__ bias,
                                 float* __restrict__ Y, int ldy,
                                 int K, int relu) {
  __shared__ __attribute__((aligned(16))) u16 sA[128][40];  // pad->conflict-free b128
  __shared__ __attribute__((aligned(16))) u16 sB[128][40];
  const int tid  = threadIdx.x;
  const int lane = tid & 31;
  const int wave = tid >> 5;
  const int wr   = wave & 3;        // 4 row-subtiles of 32
  const int wc   = wave >> 2;       // 2 col-subtiles of 64
  const int hb   = (lane >> 4) << 3;
  const int l16  = lane & 15;
  const int row0 = blockIdx.y * 128;
  const int col0 = blockIdx.x * 128;
  const int lr   = tid >> 1;        // 0..127 : staged row
  const int lk   = (tid & 1) * 16;  // 0/16  : staged k-chunk

  const v8f vzero = {};
  v8f acc[2][4];
#pragma unroll
  for (int i = 0; i < 2; ++i)
#pragma unroll
    for (int j = 0; j < 4; ++j) acc[i][j] = vzero;

  for (int k0 = 0; k0 < K; k0 += 32) {
    const float* xp = X + (size_t)(row0 + lr) * ldx + k0 + lk;
    v4f x0 = *(const v4f*)xp;
    v4f x1 = *(const v4f*)(xp + 4);
    v4f x2 = *(const v4f*)(xp + 8);
    v4f x3 = *(const v4f*)(xp + 12);
    const float* wp = W + (size_t)(col0 + lr) * ldw + k0 + lk;
    v4f w0 = *(const v4f*)wp;
    v4f w1 = *(const v4f*)(wp + 4);
    v4f w2 = *(const v4f*)(wp + 8);
    v4f w3 = *(const v4f*)(wp + 12);
    v4u32 pa0 = {pack2(x0[0], x0[1]), pack2(x0[2], x0[3]),
                 pack2(x1[0], x1[1]), pack2(x1[2], x1[3])};
    v4u32 pa1 = {pack2(x2[0], x2[1]), pack2(x2[2], x2[3]),
                 pack2(x3[0], x3[1]), pack2(x3[2], x3[3])};
    v4u32 pb0 = {pack2(w0[0], w0[1]), pack2(w0[2], w0[3]),
                 pack2(w1[0], w1[1]), pack2(w1[2], w1[3])};
    v4u32 pb1 = {pack2(w2[0], w2[1]), pack2(w2[2], w2[3]),
                 pack2(w3[0], w3[1]), pack2(w3[2], w3[3])};
    *(v4u32*)&sA[lr][lk]     = pa0;
    *(v4u32*)&sA[lr][lk + 8] = pa1;
    *(v4u32*)&sB[lr][lk]     = pb0;
    *(v4u32*)&sB[lr][lk + 8] = pb1;
    __syncthreads();
    v16bf a0 = frag_from_lds_row(&sA[wr * 32 + l16][0], hb);
    v16bf a1 = frag_from_lds_row(&sA[wr * 32 + 16 + l16][0], hb);
    v16bf b0 = frag_from_lds_row(&sB[wc * 64 + l16][0], hb);
    v16bf b1 = frag_from_lds_row(&sB[wc * 64 + 16 + l16][0], hb);
    v16bf b2 = frag_from_lds_row(&sB[wc * 64 + 32 + l16][0], hb);
    v16bf b3 = frag_from_lds_row(&sB[wc * 64 + 48 + l16][0], hb);
    acc[0][0] = __builtin_amdgcn_wmma_f32_16x16x32_bf16(false, a0, false, b0, (short)0, acc[0][0], false, false);
    acc[0][1] = __builtin_amdgcn_wmma_f32_16x16x32_bf16(false, a0, false, b1, (short)0, acc[0][1], false, false);
    acc[0][2] = __builtin_amdgcn_wmma_f32_16x16x32_bf16(false, a0, false, b2, (short)0, acc[0][2], false, false);
    acc[0][3] = __builtin_amdgcn_wmma_f32_16x16x32_bf16(false, a0, false, b3, (short)0, acc[0][3], false, false);
    acc[1][0] = __builtin_amdgcn_wmma_f32_16x16x32_bf16(false, a1, false, b0, (short)0, acc[1][0], false, false);
    acc[1][1] = __builtin_amdgcn_wmma_f32_16x16x32_bf16(false, a1, false, b1, (short)0, acc[1][1], false, false);
    acc[1][2] = __builtin_amdgcn_wmma_f32_16x16x32_bf16(false, a1, false, b2, (short)0, acc[1][2], false, false);
    acc[1][3] = __builtin_amdgcn_wmma_f32_16x16x32_bf16(false, a1, false, b3, (short)0, acc[1][3], false, false);
    __syncthreads();
  }
#pragma unroll
  for (int i = 0; i < 2; ++i) {
    const int m0 = row0 + wr * 32 + i * 16 + hb;  // C layout: VGPR r -> M=r(+8)
#pragma unroll
    for (int j = 0; j < 4; ++j) {
      const int n0 = col0 + wc * 64 + j * 16 + l16;
      const float bv = bias[n0];
#pragma unroll
      for (int r = 0; r < 8; ++r) {
        float y = acc[i][j][r] + bv;
        if (relu) y = fmaxf(y, 0.f);
        Y[(size_t)(m0 + r) * ldy + n0] = y;
      }
    }
  }
}

// --------------- Fused (flash-style) attention, one wave per 16 q-rows ------
// Q/K/V stored token-major (token = l*BATCH + b) with leading dim ldq/ldk/ldv;
// head slice at column h*32. Online softmax in fp32, WMMA for QK^T and P*V.
__launch_bounds__(32)
__global__ void attn_kernel(const float* __restrict__ Qb, int ldq,
                            const float* __restrict__ Kb, int ldk,
                            const float* __restrict__ Vb, int ldv,
                            const float* __restrict__ mask,  // (BATCH, Lq, S) or null
                            float* __restrict__ Ob, int ldo,
                            int Lq, int S, float scale) {
  __shared__ __attribute__((aligned(16))) u16 sp[16][40];
  const int lane = threadIdx.x;
  const int l16  = lane & 15;
  const int hb   = (lane >> 4) << 3;
  const int qt   = blockIdx.x * 16;
  const int b    = blockIdx.y / NHEAD;
  const int h    = blockIdx.y % NHEAD;
  const int hoff = h * HDIM;

  const float* qrow = Qb + (size_t)((qt + l16) * BATCH + b) * ldq + hoff;
  v16bf aq = frag_from_global_row(qrow, hb);

  v8f acc0 = {}; v8f acc1 = {};
  float m_i[8], l_i[8];
#pragma unroll
  for (int r = 0; r < 8; ++r) { m_i[r] = -3.0e38f; l_i[r] = 0.f; }

  for (int s0 = 0; s0 < S; s0 += 32) {
    const int t0 = s0 + l16;                 // always < S (S multiple of 16)
    const int t1 = s0 + 16 + l16;
    const int t1c = (t1 < S) ? t1 : (S - 1);
    v16bf bk0 = frag_from_global_row(Kb + (size_t)(t0  * BATCH + b) * ldk + hoff, hb);
    v16bf bk1 = frag_from_global_row(Kb + (size_t)(t1c * BATCH + b) * ldk + hoff, hb);
    v8f z0 = {}; v8f z1 = {};
    v8f lg0 = __builtin_amdgcn_wmma_f32_16x16x32_bf16(false, aq, false, bk0, (short)0, z0, false, false);
    v8f lg1 = __builtin_amdgcn_wmma_f32_16x16x32_bf16(false, aq, false, bk1, (short)0, z1, false, false);

#pragma unroll
    for (int r = 0; r < 8; ++r) {
      const int qi = qt + r + hb;
      float v0 = lg0[r] * scale;
      float v1 = lg1[r] * scale;
      if (mask) {
        v0 += mask[((size_t)b * Lq + qi) * S + t0];
        if (t1 < S) v1 += mask[((size_t)b * Lq + qi) * S + t1];
      }
      if (t1 >= S) v1 = -3.0e38f;
      float mx = fmaxf(v0, v1);
#pragma unroll
      for (int off = 1; off < 16; off <<= 1)
        mx = fmaxf(mx, __shfl_xor(mx, off, 32));  // reduce across 16-lane group
      float mn = fmaxf(m_i[r], mx);
      float p0 = __expf(v0 - mn);
      float p1 = (t1 >= S) ? 0.f : __expf(v1 - mn);
      float sm = p0 + p1;
#pragma unroll
      for (int off = 1; off < 16; off <<= 1)
        sm += __shfl_xor(sm, off, 32);
      float alpha = __expf(m_i[r] - mn);
      m_i[r] = mn;
      l_i[r] = l_i[r] * alpha + sm;
      acc0[r] *= alpha;
      acc1[r] *= alpha;
      sp[r + hb][l16]      = f2bf(p0);    // C-layout -> row-major LDS
      sp[r + hb][l16 + 16] = f2bf(p1);
    }
    __syncthreads();
    // V fragments: B[K(e)][n] = V[token s0+K(e)][hoff + n(+16)]
    v8u32 tv0, tv1;
#pragma unroll
    for (int p = 0; p < 8; ++p) {
      const int e0 = 2 * p, e1 = 2 * p + 1;
      const int kk0 = hb + ((e0 < 8) ? e0 : e0 + 8);
      const int kk1 = hb + ((e1 < 8) ? e1 : e1 + 8);
      int tk0 = s0 + kk0; if (tk0 >= S) tk0 = S - 1;   // p==0 there anyway
      int tk1 = s0 + kk1; if (tk1 >= S) tk1 = S - 1;
      const float* vp0 = Vb + (size_t)(tk0 * BATCH + b) * ldv + hoff + l16;
      const float* vp1 = Vb + (size_t)(tk1 * BATCH + b) * ldv + hoff + l16;
      tv0[p] = pack2(vp0[0], vp1[0]);
      tv1[p] = pack2(vp0[16], vp1[16]);
    }
    v16bf bv0 = __builtin_bit_cast(v16bf, tv0);
    v16bf bv1 = __builtin_bit_cast(v16bf, tv1);
    v16bf ap = frag_from_lds_row(&sp[l16][0], hb);
    acc0 = __builtin_amdgcn_wmma_f32_16x16x32_bf16(false, ap, false, bv0, (short)0, acc0, false, false);
    acc1 = __builtin_amdgcn_wmma_f32_16x16x32_bf16(false, ap, false, bv1, (short)0, acc1, false, false);
    __syncthreads();
  }
#pragma unroll
  for (int r = 0; r < 8; ++r) {
    const float inv = 1.f / l_i[r];
    float* op = Ob + (size_t)((qt + r + hb) * BATCH + b) * ldo + hoff + l16;
    op[0]  = acc0[r] * inv;
    op[16] = acc1[r] * inv;
  }
}

// ---------------- LayerNorm (optionally fused residual), D=256 --------------
__launch_bounds__(256)
__global__ void ln_kernel(const float* __restrict__ X, const float* __restrict__ R,
                          const float* __restrict__ g, const float* __restrict__ bta,
                          float* __restrict__ Y) {
  __shared__ float red[256];
  const int tid = threadIdx.x;
  const size_t row = blockIdx.x;
  float v = X[row * D_MODEL + tid];
  if (R) v += R[row * D_MODEL + tid];
  red[tid] = v; __syncthreads();
  for (int s = 128; s > 0; s >>= 1) { if (tid < s) red[tid] += red[tid + s]; __syncthreads(); }
  const float mu = red[0] * (1.0f / D_MODEL);
  __syncthreads();
  const float d = v - mu;
  red[tid] = d * d; __syncthreads();
  for (int s = 128; s > 0; s >>= 1) { if (tid < s) red[tid] += red[tid + s]; __syncthreads(); }
  const float var = red[0] * (1.0f / D_MODEL);
  Y[row * D_MODEL + tid] = d * rsqrtf(var + LN_EPSF) * g[tid] + bta[tid];
}

// ---------------- elementwise ----------------
__global__ void add_kernel(float* __restrict__ y, const float* __restrict__ a,
                           const float* __restrict__ b, int n) {
  int i = blockIdx.x * blockDim.x + threadIdx.x;
  if (i < n) y[i] = a[i] + b[i];
}
__global__ void copy_kernel(float* __restrict__ y, const float* __restrict__ a, int n) {
  int i = blockIdx.x * blockDim.x + threadIdx.x;
  if (i < n) y[i] = a[i];
}
__global__ void fill0_kernel(float* __restrict__ y, int n) {
  int i = blockIdx.x * blockDim.x + threadIdx.x;
  if (i < n) y[i] = 0.f;
}

// ---------------- host-side sequencing ----------------
static inline void gemm(hipStream_t s, const float* X, int ldx, const float* W, int ldw,
                        const float* bias, float* Y, int ldy,
                        int rows, int cols, int K, int relu) {
  dim3 g(cols / 128, rows / 128);
  gemm_bias_kernel<<<g, 256, 0, s>>>(X, ldx, W, ldw, bias, Y, ldy, K, relu);
}
static inline void ew_add(hipStream_t s, float* y, const float* a, const float* b, int n) {
  add_kernel<<<(n + 255) / 256, 256, 0, s>>>(y, a, b, n);
}

extern "C" void kernel_launch(void* const* d_in, const int* in_sizes, int n_in,
                              void* d_out, int out_size, void* d_ws, size_t ws_size,
                              hipStream_t stream) {
  (void)in_sizes; (void)n_in; (void)out_size; (void)ws_size;
  const float* src         = (const float*)d_in[0];
  const float* pos         = (const float*)d_in[1];
  const float* query_embed = (const float*)d_in[2];
  const float* cam_mask    = (const float*)d_in[3];
  const float* enc_qkv_w   = (const float*)d_in[4];
  const float* enc_qkv_b   = (const float*)d_in[5];
  const float* enc_out_w   = (const float*)d_in[6];
  const float* enc_out_b   = (const float*)d_in[7];
  const float* enc_ln1_g   = (const float*)d_in[8];
  const float* enc_ln1_b   = (const float*)d_in[9];
  const float* enc_ff1_w   = (const float*)d_in[10];
  const float* enc_ff1_b   = (const float*)d_in[11];
  const float* enc_ff2_w   = (const float*)d_in[12];
  const float* enc_ff2_b   = (const float*)d_in[13];
  const float* enc_ln2_g   = (const float*)d_in[14];
  const float* enc_ln2_b   = (const float*)d_in[15];
  const float* dsa_qkv_w   = (const float*)d_in[16];
  const float* dsa_qkv_b   = (const float*)d_in[17];
  const float* dsa_out_w   = (const float*)d_in[18];
  const float* dsa_out_b   = (const float*)d_in[19];
  const float* dln1_g      = (const float*)d_in[20];
  const float* dln1_b      = (const float*)d_in[21];
  const float* dca_qkv_w   = (const float*)d_in[22];
  const float* dca_qkv_b   = (const float*)d_in[23];
  const float* dca_out_w   = (const float*)d_in[24];
  const float* dca_out_b   = (const float*)d_in[25];
  const float* dln2_g      = (const float*)d_in[26];
  const float* dln2_b      = (const float*)d_in[27];
  const float* dff1_w      = (const float*)d_in[28];
  const float* dff1_b      = (const float*)d_in[29];
  const float* dff2_w      = (const float*)d_in[30];
  const float* dff2_b      = (const float*)d_in[31];
  const float* dln3_g      = (const float*)d_in[32];
  const float* dln3_b      = (const float*)d_in[33];
  const float* dnorm_g     = (const float*)d_in[34];
  const float* dnorm_b     = (const float*)d_in[35];

  const int NE = L_ENC * BATCH;          // 8192 encoder tokens
  const int NDQ = N_Q * BATCH;           // 640 decoder tokens
  const float scale = 0.17677669529663687f;  // 1/sqrt(32)

  float* ws = (float*)d_ws;
  float* mem  = ws;                               // 8192*256
  float* bufA = mem  + (size_t)NE * D_MODEL;      // 8192*256 (qk / attn-out)
  float* bQKV = bufA + (size_t)NE * D_MODEL;      // 8192*768
  float* bufF = bQKV + (size_t)NE * 768;          // 8192*2048 (ffn hidden / CA kv)
  float* bufP = bufF + (size_t)NE * FFDIM;        // 8192*256
  float* tgt  = bufP + (size_t)NE * D_MODEL;      // 640*256
  float* dA   = tgt  + (size_t)NDQ * D_MODEL;     // 640*256
  float* dQKV = dA   + (size_t)NDQ * D_MODEL;     // 640*768
  float* dP   = dQKV + (size_t)NDQ * 768;         // 640*256
  float* dH   = dP   + (size_t)NDQ * D_MODEL;     // 640*2048

  // mem = src
  copy_kernel<<<(NE * D_MODEL + 255) / 256, 256, 0, stream>>>(mem, src, NE * D_MODEL);

  // -------- encoder --------
  for (int i = 0; i < 2; ++i) {
    const float* wq = enc_qkv_w + (size_t)i * 768 * 256;
    const float* wk = wq + 256 * 256;
    const float* wv = wq + 512 * 256;
    const float* bq = enc_qkv_b + (size_t)i * 768;
    const float* bk = bq + 256;
    const float* bv = bq + 512;
    ew_add(stream, bufA, mem, pos, NE * D_MODEL);                         // qk = mem+pos
    gemm(stream, bufA, 256, wq, 256, bq, bQKV + 0,   768, NE, 256, 256, 0);
    gemm(stream, bufA, 256, wk, 256, bk, bQKV + 256, 768, NE, 256, 256, 0);
    gemm(stream, mem,  256, wv, 256, bv, bQKV + 512, 768, NE, 256, 256, 0);
    attn_kernel<<<dim3(L_ENC / 16, BATCH * NHEAD), 32, 0, stream>>>(
        bQKV + 0, 768, bQKV + 256, 768, bQKV + 512, 768, nullptr,
        bufA, 256, L_ENC, L_ENC, scale);
    gemm(stream, bufA, 256, enc_out_w + (size_t)i * 256 * 256, 256,
         enc_out_b + (size_t)i * 256, bufP, 256, NE, 256, 256, 0);
    ln_kernel<<<NE, 256, 0, stream>>>(mem, bufP, enc_ln1_g + i * 256, enc_ln1_b + i * 256, mem);
    gemm(stream, mem, 256, enc_ff1_w + (size_t)i * FFDIM * 256, 256,
         enc_ff1_b + (size_t)i * FFDIM, bufF, FFDIM, NE, FFDIM, 256, 1);
    gemm(stream, bufF, FFDIM, enc_ff2_w + (size_t)i * 256 * FFDIM, FFDIM,
         enc_ff2_b + (size_t)i * 256, bufP, 256, NE, 256, FFDIM, 0);
    ln_kernel<<<NE, 256, 0, stream>>>(mem, bufP, enc_ln2_g + i * 256, enc_ln2_b + i * 256, mem);
  }

  // -------- decoder --------
  fill0_kernel<<<(NDQ * D_MODEL + 255) / 256, 256, 0, stream>>>(tgt, NDQ * D_MODEL);
  for (int i = 0; i < 2; ++i) {
    // self-attention
    const float* swq = dsa_qkv_w + (size_t)i * 768 * 256;
    const float* swk = swq + 256 * 256;
    const float* swv = swq + 512 * 256;
    const float* sbq = dsa_qkv_b + (size_t)i * 768;
    ew_add(stream, dA, tgt, query_embed, NDQ * D_MODEL);                  // qk = tgt+qe
    gemm(stream, dA,  256, swq, 256, sbq,       dQKV + 0,   768, NDQ, 256, 256, 0);
    gemm(stream, dA,  256, swk, 256, sbq + 256, dQKV + 256, 768, NDQ, 256, 256, 0);
    gemm(stream, tgt, 256, swv, 256, sbq + 512, dQKV + 512, 768, NDQ, 256, 256, 0);
    attn_kernel<<<dim3(N_Q / 16, BATCH * NHEAD), 32, 0, stream>>>(
        dQKV + 0, 768, dQKV + 256, 768, dQKV + 512, 768, nullptr,
        dA, 256, N_Q, N_Q, scale);
    gemm(stream, dA, 256, dsa_out_w + (size_t)i * 256 * 256, 256,
         dsa_out_b + (size_t)i * 256, dP, 256, NDQ, 256, 256, 0);
    ln_kernel<<<NDQ, 256, 0, stream>>>(tgt, dP, dln1_g + i * 256, dln1_b + i * 256, tgt);

    // cross-attention (q from tgt+qe; k from mem+pos; v from mem; bias=cam_mask)
    const float* cwq = dca_qkv_w + (size_t)i * 768 * 256;
    const float* cwk = cwq + 256 * 256;
    const float* cwv = cwq + 512 * 256;
    const float* cbq = dca_qkv_b + (size_t)i * 768;
    ew_add(stream, dA, tgt, query_embed, NDQ * D_MODEL);
    gemm(stream, dA, 256, cwq, 256, cbq, dQKV + 0, 768, NDQ, 256, 256, 0);
    ew_add(stream, bufA, mem, pos, NE * D_MODEL);
    gemm(stream, bufA, 256, cwk, 256, cbq + 256, bufF + 0,   512, NE, 256, 256, 0);
    gemm(stream, mem,  256, cwv, 256, cbq + 512, bufF + 256, 512, NE, 256, 256, 0);
    attn_kernel<<<dim3(N_Q / 16, BATCH * NHEAD), 32, 0, stream>>>(
        dQKV + 0, 768, bufF + 0, 512, bufF + 256, 512, cam_mask,
        dA, 256, N_Q, L_ENC, scale);
    gemm(stream, dA, 256, dca_out_w + (size_t)i * 256 * 256, 256,
         dca_out_b + (size_t)i * 256, dP, 256, NDQ, 256, 256, 0);
    ln_kernel<<<NDQ, 256, 0, stream>>>(tgt, dP, dln2_g + i * 256, dln2_b + i * 256, tgt);

    // FFN
    gemm(stream, tgt, 256, dff1_w + (size_t)i * FFDIM * 256, 256,
         dff1_b + (size_t)i * FFDIM, dH, FFDIM, NDQ, FFDIM, 256, 1);
    gemm(stream, dH, FFDIM, dff2_w + (size_t)i * 256 * FFDIM, FFDIM,
         dff2_b + (size_t)i * 256, dP, 256, NDQ, 256, FFDIM, 0);
    ln_kernel<<<NDQ, 256, 0, stream>>>(tgt, dP, dln3_g + i * 256, dln3_b + i * 256, tgt);
  }

  // final norm -> d_out (fp32, (1,80,8,256))
  ln_kernel<<<NDQ, 256, 0, stream>>>(tgt, nullptr, dnorm_g, dnorm_b, (float*)d_out);
}
